// Spatial_decom_50156628083377
// MI455X (gfx1250) — matrix operand order repaired and checked by reading, not compiled, not run on hardware
//
#include <hip/hip_runtime.h>
#include <hip/hip_bf16.h>
#include <stdint.h>

#define BT   192          // B*T
#define NN   1024         // nodes
#define FF   128          // feature dim
#define HH   8            // heads
#define DHH  16           // d_head
#define CC   64           // cores
#define NTOT (BT*NN)      // rows for batchnorm stats

typedef __attribute__((ext_vector_type(16))) __bf16 v16bf;
typedef __attribute__((ext_vector_type(8)))  float  v8f;
typedef __attribute__((ext_vector_type(4)))  unsigned int u32x4;
typedef __attribute__((ext_vector_type(8)))  int          i32x8;
typedef __attribute__((ext_vector_type(4)))  int          i32x4;

union FragU { uint4 q[2]; v16bf v; };
union PackU { unsigned short s[8]; uint4 v; };

__device__ __forceinline__ unsigned short f2bf(float x){
  unsigned int u = __float_as_uint(x);
  u += 0x7fffu + ((u >> 16) & 1u);          // round-to-nearest-even
  return (unsigned short)(u >> 16);
}
__device__ __forceinline__ float bf2f(unsigned short s){
  return __uint_as_float(((unsigned int)s) << 16);
}

// A fragment (16x32 bf16, row-major, ld in elements): lanes 0-15 rows 0-15
// with K chunks {0..7,16..23}; lanes 16-31 same rows, chunks {8..15,24..31}.
__device__ __forceinline__ v16bf frag_a(const unsigned short* base, int ld, int k0){
  const int lane = threadIdx.x & 31;
  const unsigned short* p = base + (size_t)(lane & 15) * ld + k0 + ((lane >> 4) << 3);
  FragU f;
  f.q[0] = *(const uint4*)(p);
  f.q[1] = *(const uint4*)(p + 16);
  return f.v;
}
// B fragment (32x16 bf16) from column-major storage (column = out index,
// contiguous K): lanes 0-15 column n, K {0..7,16..23}; lanes 16-31 K {8..15,24..31}.
__device__ __forceinline__ v16bf frag_b(const unsigned short* base, int ld, int k0){
  const int lane = threadIdx.x & 31;
  const unsigned short* p = base + (size_t)(lane & 15) * ld + k0 + ((lane >> 4) << 3);
  FragU f;
  f.q[0] = *(const uint4*)(p);
  f.q[1] = *(const uint4*)(p + 16);
  return f.v;
}
__device__ __forceinline__ v8f wmma_bf(v16bf a, v16bf b, v8f c){
  return __builtin_amdgcn_wmma_f32_16x16x32_bf16(false, a, false, b, (short)0, c, false, false);
}

// ---- Tensor Data Mover: 1D contiguous global->LDS copy (dword granularity) ----
// D# per CDNA5 ISA ch.8: 2D tile of one row, tile_dim0 = `dwords` 4B elements.
// LDS byte address = low 32 bits of the generic shared-memory pointer (ISA 10.2).
// This toolchain declares the 6-arg builtin form (extra int32x8 group; zeros ->
// tile_dim3/4 = 0 = unused for a <=2D tile).
__device__ __forceinline__ void tdm_load_1d(void* lds, const void* gsrc, unsigned dwords){
  const unsigned lds_addr = (unsigned)(uintptr_t)lds;
  const unsigned long long ga = (unsigned long long)(uintptr_t)gsrc;
  u32x4 g0;
  g0[0] = 1u;                                            // count=1, user mode
  g0[1] = lds_addr;                                      // lds_addr
  g0[2] = (unsigned)(ga & 0xffffffffu);                  // global_addr[31:0]
  g0[3] = (unsigned)((ga >> 32) & 0x01ffffffu) | 0x80000000u; // addr[56:32] | type=2
  i32x8 g1;
  g1[0] = (int)(2u << 16);                               // mask=0, data_size=4B
  g1[1] = (int)((dwords & 0xffffu) << 16);               // tensor_dim0[15:0]
  g1[2] = (int)(((dwords >> 16) & 0xffffu) | (1u << 16));// dim0[31:16] | tensor_dim1=1
  g1[3] = (int)((dwords & 0xffffu) << 16);               // tile_dim0 (16-bit)
  g1[4] = 1;                                             // tile_dim1=1, tile_dim2=0
  g1[5] = (int)dwords;                                   // tensor_dim0_stride[31:0]
  g1[6] = (int)((dwords & 0xffffu) << 16);               // stride0[47:32]=0 | stride1[15:0]
  g1[7] = (int)(dwords >> 16);                           // stride1[47:16]
  const i32x4 z4 = {0, 0, 0, 0};
  const i32x8 z8 = {0, 0, 0, 0, 0, 0, 0, 0};
  __builtin_amdgcn_tensor_load_to_lds(g0, g1, z4, z4, z8, 0);
}

// ---------------- prep: bf16 transposed weights + zero BN stats ----------------
__global__ void __launch_bounds__(256) prep_kernel(
    const float* __restrict__ W_v, const float* __restrict__ W1,
    const float* __restrict__ W2, unsigned short* __restrict__ wvt,
    unsigned short* __restrict__ w1t, unsigned short* __restrict__ w2t,
    float* __restrict__ stats){
  const int i = blockIdx.x * 256 + threadIdx.x;       // grid covers 262144
  if (i < 128*128){ int o = i >> 7,  k = i & 127;  wvt[o*128 + k]  = f2bf(W_v[(size_t)k*128  + o]); }
  if (i < 1024*256){ int o = i >> 8,  k = i & 255;  w1t[o*256 + k]  = f2bf(W1[(size_t)k*1024 + o]); }
  if (i < 128*1024){ int o = i >> 10, k = i & 1023; w2t[o*1024 + k] = f2bf(W2[(size_t)k*128  + o]); }
  if (i < 512) stats[i] = 0.0f;
}

// ---------------- affinity scores aff[h][c][n] ----------------
__global__ void __launch_bounds__(256) aff_kernel(
    const float* __restrict__ cores, const float* __restrict__ adaptive,
    float* __restrict__ aff){
  const int i = blockIdx.x * 256 + threadIdx.x;       // 8*64*1024 total
  const int n  = i & 1023;
  const int hc = i >> 10;
  const int h  = hc >> 6;
  const int c  = hc & 63;
  const float* cp = cores + (size_t)(h*64 + c) * 16;
  const float* ap = adaptive + (size_t)h*16*1024 + n;
  float s = 0.0f;
#pragma unroll
  for (int d = 0; d < 16; ++d) s += cp[d] * ap[(size_t)d*1024];
  aff[i] = s * 0.25f;                                  // 1/sqrt(16)
}

// softmax over n -> a1[h][c][n] bf16 (B-matrix layout: column c, contiguous k=n)
__global__ void __launch_bounds__(256) sm_n_kernel(
    const float* __restrict__ aff, unsigned short* __restrict__ a1){
  __shared__ float red[256];
  const float* row = aff + (size_t)blockIdx.x * 1024;
  float m = -1e30f;
  for (int i = threadIdx.x; i < 1024; i += 256) m = fmaxf(m, row[i]);
  red[threadIdx.x] = m; __syncthreads();
  for (int s = 128; s > 0; s >>= 1){
    if (threadIdx.x < s) red[threadIdx.x] = fmaxf(red[threadIdx.x], red[threadIdx.x + s]);
    __syncthreads();
  }
  m = red[0]; __syncthreads();
  float sum = 0.0f;
  for (int i = threadIdx.x; i < 1024; i += 256) sum += __expf(row[i] - m);
  red[threadIdx.x] = sum; __syncthreads();
  for (int s = 128; s > 0; s >>= 1){
    if (threadIdx.x < s) red[threadIdx.x] += red[threadIdx.x + s];
    __syncthreads();
  }
  const float inv = 1.0f / red[0];
  unsigned short* orow = a1 + (size_t)blockIdx.x * 1024;
  for (int i = threadIdx.x; i < 1024; i += 256) orow[i] = f2bf(__expf(row[i] - m) * inv);
}

// softmax over c -> a2t[h][n][c] bf16 (B-matrix layout: column n, contiguous k=c)
__global__ void __launch_bounds__(256) sm_c_kernel(
    const float* __restrict__ aff, unsigned short* __restrict__ a2t){
  const int i = blockIdx.x * 256 + threadIdx.x;        // 8*1024 total
  const int h = i >> 10;
  const int n = i & 1023;
  const float* col = aff + (size_t)h*64*1024 + n;
  float m = -1e30f;
  for (int c = 0; c < 64; ++c) m = fmaxf(m, col[(size_t)c*1024]);
  float sum = 0.0f;
  for (int c = 0; c < 64; ++c) sum += __expf(col[(size_t)c*1024] - m);
  const float inv = 1.0f / sum;
  unsigned short* o = a2t + (size_t)i * 64;
  for (int c = 0; c < 64; ++c) o[c] = f2bf(__expf(col[(size_t)c*1024] - m) * inv);
}

// ------- value projection: vhead[bt][h][dh][n] = (in@Wv+b)^T, 32-row tiles -------
__global__ void __launch_bounds__(256) wv_kernel(
    const float* __restrict__ input, const float* __restrict__ b_v,
    const unsigned short* __restrict__ wvt, unsigned short* __restrict__ vhead){
  __shared__ __align__(16) float          s_in[32*128];  // 16KB TDM staging
  __shared__ __align__(16) unsigned short xa[32*128];    //  8KB bf16 A tile
  const int bt = blockIdx.x >> 5;
  const int n0 = (blockIdx.x & 31) << 5;
  const int tid = threadIdx.x;
  if (tid < 32){                                          // wave 0 issues the DMA
    tdm_load_1d(s_in, input + ((size_t)bt*NN + n0)*FF, 32*128);
    __builtin_amdgcn_s_wait_tensorcnt(0);
  }
  __syncthreads();
  for (int i = tid; i < 32*128; i += 256) xa[i] = f2bf(s_in[i]);
  __syncthreads();
  const int h    = tid >> 5;                              // wave = head
  const int lane = tid & 31;
  v8f c0 = {}, c1 = {};
#pragma unroll
  for (int ks = 0; ks < 4; ++ks){
    v16bf b = frag_b(wvt + (size_t)(h*16)*128, 128, ks*32);   // one B, two WMMAs
    c0 = wmma_bf(frag_a(xa,          128, ks*32), b, c0);
    c1 = wmma_bf(frag_a(xa + 16*128, 128, ks*32), b, c1);
  }
  const int dh   = lane & 15;
  const float bias = b_v[h*16 + dh];
  const int rb   = (lane >> 4) << 3;                      // node-row base 0 or 8
  PackU p0, p1;
#pragma unroll
  for (int r = 0; r < 8; ++r){ p0.s[r] = f2bf(c0[r] + bias); p1.s[r] = f2bf(c1[r] + bias); }
  unsigned short* dst = vhead + (((size_t)bt*HH + h)*DHH + dh)*NN + n0 + rb;
  *(uint4*)dst        = p0.v;                             // n0+rb   .. +7
  *(uint4*)(dst + 16) = p1.v;                             // n0+16+rb.. +7
}

// ---------------- two-stage core aggregation per (bt, h) ----------------
__global__ void __launch_bounds__(128) core_kernel(
    const unsigned short* __restrict__ vhead, const unsigned short* __restrict__ a1,
    const unsigned short* __restrict__ a2t, unsigned short* __restrict__ vmerged){
  __shared__ __align__(16) unsigned short tmp[16*64];
  const int bt = blockIdx.x >> 3;
  const int h  = blockIdx.x & 7;
  const int w    = threadIdx.x >> 5;       // 4 waves
  const int lane = threadIdx.x & 31;
  const int rb   = (lane >> 4) << 3;
  // stage A: tmp[dh][c] = sum_n v[dh][n] * a1[c][n]   (K = 1024)
  const unsigned short* Ab = vhead + ((size_t)bt*HH + h)*DHH*NN;
  const unsigned short* Bb = a1 + ((size_t)h*CC + w*16)*NN;
  v8f c = {};
  for (int ks = 0; ks < 32; ++ks)
    c = wmma_bf(frag_a(Ab, NN, ks*32), frag_b(Bb, NN, ks*32), c);
  const int col = w*16 + (lane & 15);
#pragma unroll
  for (int r = 0; r < 8; ++r) tmp[(size_t)(r + rb)*64 + col] = f2bf(c[r]);
  __syncthreads();
  // stage B: out[dh][n] = sum_c tmp[dh][c] * a2t[n][c]   (K = 64)
  for (int t = 0; t < 16; ++t){
    const int n0 = (w*16 + t) * 16;
    const unsigned short* B2 = a2t + ((size_t)h*NN + n0)*CC;
    v8f d = {};
    d = wmma_bf(frag_a(tmp, 64, 0),  frag_b(B2, 64, 0),  d);
    d = wmma_bf(frag_a(tmp, 64, 32), frag_b(B2, 64, 32), d);
    const int nc = n0 + (lane & 15);
    PackU pk;
#pragma unroll
    for (int r = 0; r < 8; ++r) pk.s[r] = f2bf(d[r]);   // dh = rb..rb+7 contiguous in f
    unsigned short* dst = vmerged + ((size_t)bt*NN + nc)*FF + h*16 + rb;
    *(uint4*)dst = pk.v;
  }
}

// ---------------- fused FFN + residual, 32-row tiles ----------------
__global__ void __launch_bounds__(256) ffn_kernel(
    const float* __restrict__ input, const unsigned short* __restrict__ vmerged,
    const unsigned short* __restrict__ w1t, const float* __restrict__ b1,
    const unsigned short* __restrict__ w2t, const float* __restrict__ b2,
    float* __restrict__ out){
  __shared__ __align__(16) unsigned short xt[32*256];    // 16 KB, x = [in-v, v]
  __shared__ __align__(16) unsigned short h1s[32*1024];  // 64 KB, gelu output
  __shared__ __align__(16) float          s_in[32*128];  // 16 KB, input staging
  unsigned short* s_vm = h1s;                            // 8 KB alias (dead after x built)
  const int bt = blockIdx.x >> 5;
  const int n0 = (blockIdx.x & 31) << 5;
  const int tid = threadIdx.x;
  const size_t rowbase = ((size_t)bt*NN + n0) * FF;
  if (tid < 32){                                          // wave 0 issues both DMAs
    tdm_load_1d(s_in, input + rowbase, 32*128);           // 16 KB f32
    tdm_load_1d(s_vm, vmerged + rowbase, (32*128) / 2);   //  8 KB bf16 (as dwords)
    __builtin_amdgcn_s_wait_tensorcnt(0);
  }
  __syncthreads();
  for (int i = tid; i < 32*128; i += 256){
    const int r = i >> 7, f = i & 127;
    const float vv = bf2f(s_vm[i]);
    xt[r*256 + f]       = f2bf(s_in[i] - vv);
    xt[r*256 + 128 + f] = f2bf(vv);
  }
  __syncthreads();
  const int w    = tid >> 5;
  const int lane = tid & 31;
  const int rb   = (lane >> 4) << 3;
  // stage 1: h1 = gelu(x @ W1 + b1), K = 256; one B fragment feeds two M-tiles
  for (int t = 0; t < 8; ++t){
    const int o0 = (w*8 + t) * 16;
    v8f c0 = {}, c1 = {};
#pragma unroll
    for (int ks = 0; ks < 8; ++ks){
      v16bf b = frag_b(w1t + (size_t)o0*256, 256, ks*32);
      c0 = wmma_bf(frag_a(xt,          256, ks*32), b, c0);
      c1 = wmma_bf(frag_a(xt + 16*256, 256, ks*32), b, c1);
    }
    const int col = o0 + (lane & 15);
    const float bias = b1[col];
#pragma unroll
    for (int r = 0; r < 8; ++r){
      const float y0 = c0[r] + bias;
      const float y1 = c1[r] + bias;
      h1s[(size_t)(r + rb)*1024 + col]      = f2bf(0.5f*y0*(1.0f + erff(y0*0.70710678f)));
      h1s[(size_t)(r + rb + 16)*1024 + col] = f2bf(0.5f*y1*(1.0f + erff(y1*0.70710678f)));
    }
  }
  __syncthreads();
  // stage 2: out = h1 @ W2 + b2 + input, K = 1024
  {
    const int o0 = w * 16;
    v8f c0 = {}, c1 = {};
    for (int ks = 0; ks < 32; ++ks){
      v16bf b = frag_b(w2t + (size_t)o0*1024, 1024, ks*32);
      c0 = wmma_bf(frag_a(h1s,           1024, ks*32), b, c0);
      c1 = wmma_bf(frag_a(h1s + 16*1024, 1024, ks*32), b, c1);
    }
    const int col = o0 + (lane & 15);
    const float bias = b2[col];
#pragma unroll
    for (int r = 0; r < 8; ++r){
      const int row = r + rb;
      out[rowbase + (size_t)row*FF + col]        = c0[r] + bias + s_in[row*128 + col];
      out[rowbase + (size_t)(row + 16)*FF + col] = c1[r] + bias + s_in[(row + 16)*128 + col];
    }
  }
}

// ---------------- batchnorm: stats, finalize, normalize ----------------
__global__ void __launch_bounds__(256) bnstats_kernel(
    const float* __restrict__ out, float* __restrict__ stats){
  __shared__ float r1[256], r2[256];
  const int c    = threadIdx.x & 127;
  const int half = threadIdx.x >> 7;
  float s1 = 0.0f, s2 = 0.0f;
  const float* base = out + (size_t)blockIdx.x * NN * FF + c;
  for (int r = half; r < NN; r += 2){
    const float v = base[(size_t)r * FF];
    s1 += v; s2 += v * v;
  }
  r1[threadIdx.x] = s1; r2[threadIdx.x] = s2;
  __syncthreads();
  if (threadIdx.x < 128){
    atomicAdd(&stats[c],       r1[threadIdx.x] + r1[threadIdx.x + 128]);
    atomicAdd(&stats[128 + c], r2[threadIdx.x] + r2[threadIdx.x + 128]);
  }
}

__global__ void bnfin_kernel(float* __restrict__ stats,
                             const float* __restrict__ gamma,
                             const float* __restrict__ beta){
  const int c = threadIdx.x;
  if (c < 128){
    const float inv  = 1.0f / (float)NTOT;
    const float mean = stats[c] * inv;
    const float var  = stats[128 + c] * inv - mean * mean;
    const float sc   = gamma[c] * rsqrtf(var + 1e-5f);
    stats[256 + c] = sc;
    stats[384 + c] = beta[c] - mean * sc;
  }
}

__global__ void __launch_bounds__(256) bnnorm_kernel(
    float* __restrict__ out, const float* __restrict__ stats){
  const size_t i = (size_t)blockIdx.x * 256 + threadIdx.x;  // 25165824 total, exact
  const int c = (int)(i & 127);
  out[i] = out[i] * stats[256 + c] + stats[384 + c];
}

extern "C" void kernel_launch(void* const* d_in, const int* in_sizes, int n_in,
                              void* d_out, int out_size, void* d_ws, size_t ws_size,
                              hipStream_t stream) {
  const float* input    = (const float*)d_in[0];
  const float* W_v      = (const float*)d_in[1];
  const float* b_v      = (const float*)d_in[2];
  const float* adaptive = (const float*)d_in[3];
  const float* cores    = (const float*)d_in[4];
  const float* W1       = (const float*)d_in[5];
  const float* b1       = (const float*)d_in[6];
  const float* W2       = (const float*)d_in[7];
  const float* b2       = (const float*)d_in[8];
  const float* gamma    = (const float*)d_in[9];
  const float* beta     = (const float*)d_in[10];
  float* out = (float*)d_out;

  char* ws = (char*)d_ws;
  size_t off = 0;
  auto carve = [&](size_t bytes) -> char* {
    char* p = ws + off;
    off = (off + bytes + 255) & ~(size_t)255;
    return p;
  };
  unsigned short* wvt   = (unsigned short*)carve((size_t)128*128*2);
  unsigned short* w1t   = (unsigned short*)carve((size_t)1024*256*2);
  unsigned short* w2t   = (unsigned short*)carve((size_t)128*1024*2);
  unsigned short* a1    = (unsigned short*)carve((size_t)HH*CC*NN*2);
  unsigned short* a2t   = (unsigned short*)carve((size_t)HH*NN*CC*2);
  float*          aff   = (float*)carve((size_t)HH*CC*NN*4);
  float*          stats = (float*)carve((size_t)512*4);
  unsigned short* vhead = (unsigned short*)carve((size_t)BT*HH*DHH*NN*2);
  unsigned short* vmg   = (unsigned short*)carve((size_t)BT*NN*FF*2);

  prep_kernel   <<<1024,  256, 0, stream>>>(W_v, W1, W2, wvt, w1t, w2t, stats);
  aff_kernel    <<<2048,  256, 0, stream>>>(cores, adaptive, aff);
  sm_n_kernel   <<<512,   256, 0, stream>>>(aff, a1);
  sm_c_kernel   <<<32,    256, 0, stream>>>(aff, a2t);
  wv_kernel     <<<BT*32, 256, 0, stream>>>(input, b_v, wvt, vhead);
  core_kernel   <<<BT*8,  128, 0, stream>>>(vhead, a1, a2t, vmg);
  ffn_kernel    <<<BT*32, 256, 0, stream>>>(input, vmg, w1t, b1, w2t, b2, out);
  bnstats_kernel<<<BT,    256, 0, stream>>>(out, stats);
  bnfin_kernel  <<<1,     128, 0, stream>>>(stats, gamma, beta);
  bnnorm_kernel <<<98304, 256, 0, stream>>>(out, stats);
}
